// CosformerSelfAttention_86079734546438
// MI455X (gfx1250) — matrix operand order repaired and checked by reading, not compiled, not run on hardware
//
#include <hip/hip_runtime.h>
#include <hip/hip_bf16.h>
#include <stdint.h>

// Problem constants (match reference)
constexpr int B_  = 4;
constexpr int S_  = 4096;
constexpr int D_  = 1024;
constexpr int H_  = 16;
constexpr int DH_ = 64;
constexpr int BS_ = B_ * S_;
constexpr float EPS_ = 1e-4f;

typedef __attribute__((ext_vector_type(16))) _Float16 v16h;
typedef __attribute__((ext_vector_type(8)))  float    v8f;
typedef __attribute__((ext_vector_type(4)))  unsigned int u32x4;
typedef __attribute__((ext_vector_type(8)))  int          i32x8;
typedef __attribute__((ext_vector_type(4)))  int          i32x4;

#define WMMA_F16(A, Bm, C) \
    __builtin_amdgcn_wmma_f32_16x16x32_f16(false, (A), false, (Bm), (short)0, (C), false, false)

// Tensor Data Mover availability / arity (ROCm7.2 clang-22 = 5 args,
// amdgpu-toolchain clang-23 = 6 args; see CDNA5_HIP.md)
#if defined(__has_builtin)
#if __has_builtin(__builtin_amdgcn_tensor_load_to_lds)
#define HAVE_TDM 1
#endif
#endif
#ifndef HAVE_TDM
#define HAVE_TDM 0
#endif

union FragU {
    v16h v;
    uint4 u[2];
    _Float16 h[16];
};

// A-frag: halves 0..7 = row[hi*8 .. +7], halves 8..15 = row[16+hi*8 .. +7]
__device__ __forceinline__ v16h frag_a(const _Float16* __restrict__ row, int hi) {
    FragU f;
    f.u[0] = *(const uint4*)(row + hi * 8);
    f.u[1] = *(const uint4*)(row + 16 + hi * 8);
    return f.v;
}
// B-frag: halves 0..7 = row[hi*8 .. +7], halves 8..15 = row[16+hi*8 .. +7] (row k-contiguous / transposed)
__device__ __forceinline__ v16h frag_b(const _Float16* __restrict__ row, int hi) {
    FragU f;
    f.u[0] = *(const uint4*)(row + hi * 8);
    f.u[1] = *(const uint4*)(row + 16 + hi * 8);
    return f.v;
}

// ---------------------------------------------------------------------------
// Kernel 0: per-batch mask length -> cos/sin weight tables.  grid=B_, blk=256
// ---------------------------------------------------------------------------
__global__ void cosformer_theta_kernel(const unsigned char* __restrict__ mask,
                                       float* __restrict__ cosw,
                                       float* __restrict__ sinw) {
    __shared__ int sdata[256];
    const int b   = blockIdx.x;
    const int tid = threadIdx.x;
    int acc = 0;
    for (int s = tid; s < S_; s += 256) acc += (int)mask[b * S_ + s];
    sdata[tid] = acc;
    __syncthreads();
    for (int off = 128; off > 0; off >>= 1) {
        if (tid < off) sdata[tid] += sdata[tid + off];
        __syncthreads();
    }
    const float Mb = (float)max(sdata[0], 1);
    const float c  = 3.14159265358979323846f / (2.0f * Mb);
    for (int s = tid; s < S_; s += 256) {
        float th = c * (float)s;
        cosw[b * S_ + s] = cosf(th);
        sinw[b * S_ + s] = sinf(th);
    }
}

// ---------------------------------------------------------------------------
// GEMM:  out[M,N] = act(A[M,K] @ W[K,N] + bias)
// M=16384, K=N=1024.  128x64 tile / 256 threads (8 waves), K chunk = 32.
// A staged row-major (frag rows contiguous), W staged TRANSPOSED (Bt[n][k])
// so both operands load as ds_load_b128 pairs.
// When A is f16 (ATTN @ Wo path), the A tile is DMA'd into LDS by the
// Tensor Data Mover; TDM padding (interval=16 DWORDs, amount=4 DWORDs)
// reproduces the 40-half padded row layout exactly.
// ---------------------------------------------------------------------------
constexpr int LDA_ = 40;   // padded LDS row length (halves); 80B, 16B aligned

#if HAVE_TDM
__device__ __forceinline__ void tdm_load_tile_f16(unsigned int lds_byte_off,
                                                  const _Float16* gsrc) {
    const unsigned long long ga = (unsigned long long)(uintptr_t)gsrc;
    u32x4 g0;
    g0[0] = 1u;                                    // count=1, user descriptor
    g0[1] = lds_byte_off;                          // lds_addr
    g0[2] = (unsigned int)(ga & 0xFFFFFFFFu);      // global_addr[31:0]
    g0[3] = (unsigned int)((ga >> 32) & 0x1FFFFFFu) | (2u << 30);  // addr[56:32] | type=2
    i32x8 g1;
    // data_size=2B(code1), pad_enable, pad_interval=16 DW(code3), pad_amount=4 DW(code3)
    g1[0] = (int)((1u << 16) | (1u << 20) | (3u << 22) | (3u << 25));
    g1[1] = (int)(((unsigned)D_ & 0xFFFFu) << 16);                 // tensor_dim0 lo16
    g1[2] = (int)((((unsigned)D_ >> 16) & 0xFFFFu) |
                  (((unsigned)BS_ & 0xFFFFu) << 16));              // dim0 hi | dim1 lo
    g1[3] = (int)((((unsigned)BS_ >> 16) & 0xFFFFu) | (32u << 16)); // dim1 hi | tile_dim0=32
    g1[4] = 128;                                   // tile_dim1 = 128 rows
    g1[5] = D_;                                    // tensor_dim0_stride lo32
    g1[6] = 0;
    g1[7] = 0;
    const i32x4 z4 = (i32x4){0, 0, 0, 0};
#if defined(__clang_major__) && (__clang_major__ >= 23)
    const i32x8 z8 = (i32x8){0, 0, 0, 0, 0, 0, 0, 0};
    __builtin_amdgcn_tensor_load_to_lds(g0, g1, z4, z4, z8, 0);
#else
    __builtin_amdgcn_tensor_load_to_lds(g0, g1, z4, z4, 0);
#endif
}
#endif

template <bool A_F16, bool OUT_F16>
__global__ void cosformer_gemm_kernel(const void* __restrict__ Aptr,
                                      const float* __restrict__ W,
                                      const float* __restrict__ bias,
                                      void* __restrict__ outp,
                                      const unsigned char* __restrict__ maskp,
                                      int do_relu) {
    __shared__ _Float16 As[128 * LDA_];   // 128 rows x 32 k (padded)
    __shared__ _Float16 Bt[64 * LDA_];    // 64 n-rows x 32 k (padded), transposed W

    const int nblk = blockIdx.x & 15;     // N/64 = 16
    const int mblk = blockIdx.x >> 4;
    const int m0 = mblk * 128;
    const int n0 = nblk * 64;

    const int tid  = threadIdx.x;
    const int wave = tid >> 5;            // 0..7 -> 16 rows each
    const int lane = tid & 31;
    const int lm   = lane & 15;
    const int hi   = lane >> 4;

    v8f acc[4];
#pragma unroll
    for (int j = 0; j < 4; ++j) acc[j] = (v8f){0.f, 0.f, 0.f, 0.f, 0.f, 0.f, 0.f, 0.f};

    const float*    Af = (const float*)Aptr;
    const _Float16* Ah = (const _Float16*)Aptr;

    // staging assignments (fixed per thread)
    const int ar = tid >> 1;              // A row 0..127
    const int ac = (tid & 1) * 16;        // A col base 0 / 16
    const int wr = tid >> 3;              // W k-row 0..31
    const int wc = (tid & 7) * 8;         // W col base, 8 wide

    for (int k0 = 0; k0 < D_; k0 += 32) {
        // ---- Stage A tile (128 x 32) ----
        if (A_F16) {
#if HAVE_TDM
            if (tid < 32) {  // one wave issues the DMA (TDM is per-instruction, EXEC ignored)
                tdm_load_tile_f16((unsigned int)(uintptr_t)&As[0],
                                  &Ah[(size_t)m0 * D_ + k0]);
            }
#else
            {
                _Float16* adst = &As[ar * LDA_ + ac];
                const uint4* src = (const uint4*)&Ah[(m0 + ar) * D_ + k0 + ac];
                ((uint4*)adst)[0] = src[0];
                ((uint4*)adst)[1] = src[1];
            }
#endif
        } else {
            _Float16* adst = &As[ar * LDA_ + ac];
            const float4* src = (const float4*)&Af[(m0 + ar) * D_ + k0 + ac];
            FragU tb;
#pragma unroll
            for (int q = 0; q < 4; ++q) {
                float4 x = src[q];
                tb.h[q * 4 + 0] = (_Float16)x.x;
                tb.h[q * 4 + 1] = (_Float16)x.y;
                tb.h[q * 4 + 2] = (_Float16)x.z;
                tb.h[q * 4 + 3] = (_Float16)x.w;
            }
            ((uint4*)adst)[0] = tb.u[0];
            ((uint4*)adst)[1] = tb.u[1];
        }
        // ---- Stage W tile (32 x 64) transposed into Bt[n][k] ----
        {
            const float4* src = (const float4*)&W[(k0 + wr) * D_ + n0 + wc];
#pragma unroll
            for (int q = 0; q < 2; ++q) {
                float4 x = src[q];
                Bt[(wc + q * 4 + 0) * LDA_ + wr] = (_Float16)x.x;
                Bt[(wc + q * 4 + 1) * LDA_ + wr] = (_Float16)x.y;
                Bt[(wc + q * 4 + 2) * LDA_ + wr] = (_Float16)x.z;
                Bt[(wc + q * 4 + 3) * LDA_ + wr] = (_Float16)x.w;
            }
        }
        if (k0 + 32 < D_) {
            __builtin_prefetch(&W[(k0 + 32) * D_ + n0], 0, 1);
            if (!A_F16) __builtin_prefetch(&Af[(m0 + ar) * D_ + k0 + 32], 0, 1);
        }
#if HAVE_TDM
        if (A_F16 && tid < 32) __builtin_amdgcn_s_wait_tensorcnt(0);
#endif
        __syncthreads();

        const v16h a = frag_a(&As[(wave * 16 + lm) * LDA_], hi);
#pragma unroll
        for (int j = 0; j < 4; ++j) {
            const v16h b = frag_b(&Bt[(j * 16 + lm) * LDA_], hi);
            acc[j] = WMMA_F16(a, b, acc[j]);
        }
        __syncthreads();
    }

    // Epilogue
#pragma unroll
    for (int j = 0; j < 4; ++j) {
#pragma unroll
        for (int r = 0; r < 8; ++r) {
            const int gm = m0 + wave * 16 + hi * 8 + r;
            const int gn = n0 + j * 16 + lm;
            float v = acc[j][r] + bias[gn];
            if (OUT_F16) {
                if (do_relu) v = fmaxf(v, 0.0f);
                if (maskp)   v *= (float)maskp[gm];
                ((_Float16*)outp)[gm * D_ + gn] = (_Float16)v;
            } else {
                ((float*)outp)[gm * D_ + gn] = v;
            }
        }
    }
}

// ---------------------------------------------------------------------------
// Kernel 2: per-(b,h) moment matrices  kv_cos/kv_sin [64x64] = K~^T @ V
// grid = B_*H_ = 64 blocks, 512 threads (16 waves).  K~ staged transposed
// (dh-major) so A-frags are contiguous; V staged transposed for B-frags.
// ---------------------------------------------------------------------------
__global__ void cosformer_kv_kernel(const _Float16* __restrict__ Kp,
                                    const _Float16* __restrict__ Vp,
                                    const float* __restrict__ cosw,
                                    const float* __restrict__ sinw,
                                    float* __restrict__ kvc_out,
                                    float* __restrict__ kvs_out) {
    __shared__ _Float16 kcT[64 * LDA_];   // [dh][s_local]
    __shared__ _Float16 ksT[64 * LDA_];
    __shared__ _Float16 vvT[64 * LDA_];   // [dh][s_local]

    const int bh = blockIdx.x;
    const int b  = bh / H_;
    const int hd = bh % H_;

    const int tid  = threadIdx.x;
    const int wave = tid >> 5;
    const int lane = tid & 31;
    const int lm   = lane & 15;
    const int hi   = lane >> 4;
    const int ti   = wave >> 2;
    const int tj   = wave & 3;

    v8f acc_c = (v8f){0.f, 0.f, 0.f, 0.f, 0.f, 0.f, 0.f, 0.f};
    v8f acc_s = (v8f){0.f, 0.f, 0.f, 0.f, 0.f, 0.f, 0.f, 0.f};

    for (int sc = 0; sc < S_; sc += 32) {
        for (int idx = tid; idx < 32 * 64; idx += 512) {
            const int r = idx >> 6, c = idx & 63;        // r = s_local, c = dh
            const int s  = sc + r;
            const int gi = (b * S_ + s) * D_ + hd * DH_ + c;
            const float kf = (float)Kp[gi];
            const float cw = cosw[b * S_ + s];
            const float sw = sinw[b * S_ + s];
            kcT[c * LDA_ + r] = (_Float16)(kf * cw);
            ksT[c * LDA_ + r] = (_Float16)(kf * sw);
            vvT[c * LDA_ + r] = Vp[gi];
        }
        __syncthreads();

        const v16h a_c = frag_a(&kcT[(ti * 16 + lm) * LDA_], hi);
        const v16h a_s = frag_a(&ksT[(ti * 16 + lm) * LDA_], hi);
        const v16h bfr = frag_b(&vvT[(tj * 16 + lm) * LDA_], hi);
        acc_c = WMMA_F16(a_c, bfr, acc_c);
        acc_s = WMMA_F16(a_s, bfr, acc_s);
        __syncthreads();
    }

#pragma unroll
    for (int r = 0; r < 8; ++r) {
        const int row = ti * 16 + hi * 8 + r;
        const int col = tj * 16 + lm;
        kvc_out[(bh * DH_ + row) * DH_ + col] = acc_c[r];
        kvs_out[(bh * DH_ + row) * DH_ + col] = acc_s[r];
    }
}

// ---------------------------------------------------------------------------
// Kernel 3: denominator sums  ksum_cos/ksum_sin [B,H,DH]. grid=B_*H_, blk=64
// ---------------------------------------------------------------------------
__global__ void cosformer_ksum_kernel(const _Float16* __restrict__ Kp,
                                      const float* __restrict__ cosw,
                                      const float* __restrict__ sinw,
                                      float* __restrict__ ksc,
                                      float* __restrict__ kss) {
    const int bh = blockIdx.x;
    const int b  = bh / H_;
    const int hd = bh % H_;
    const int d  = threadIdx.x;
    float ac = 0.f, as = 0.f;
    for (int s = 0; s < S_; ++s) {
        const float kf = (float)Kp[(b * S_ + s) * D_ + hd * DH_ + d];
        ac += kf * cosw[b * S_ + s];
        as += kf * sinw[b * S_ + s];
    }
    ksc[bh * DH_ + d] = ac;
    kss[bh * DH_ + d] = as;
}

// ---------------------------------------------------------------------------
// Kernel 4: attn = (q_cos @ kv_cos + q_sin @ kv_sin) / denom
// grid = B_*H_*(S_/128) = 2048 blocks, 256 threads (8 waves x 16 rows).
// kv matrices staged transposed (n-major) so B-frags are b128 loads.
// ---------------------------------------------------------------------------
constexpr int LDK_ = 72;   // padded row length for 64-wide k rows (144B, 16B aligned)

__global__ void cosformer_attn_kernel(const _Float16* __restrict__ Qp,
                                      const float* __restrict__ kvc_in,
                                      const float* __restrict__ kvs_in,
                                      const float* __restrict__ ksc_in,
                                      const float* __restrict__ kss_in,
                                      const float* __restrict__ cosw,
                                      const float* __restrict__ sinw,
                                      _Float16* __restrict__ attn) {
    __shared__ _Float16 kvcT[64 * LDK_];   // [n][k]
    __shared__ _Float16 kvsT[64 * LDK_];
    __shared__ float ksc[64];
    __shared__ float kss[64];

    const int sblk = blockIdx.x & 31;      // S_/128 = 32
    const int bh   = blockIdx.x >> 5;
    const int b    = bh / H_;
    const int hd   = bh % H_;
    const int sbase = sblk * 128;

    const int tid  = threadIdx.x;
    const int wave = tid >> 5;
    const int lane = tid & 31;
    const int lm   = lane & 15;
    const int hi   = lane >> 4;

    for (int idx = tid; idx < 64 * 64; idx += 256) {
        const int r = idx >> 6, c = idx & 63;          // r = k, c = n
        kvcT[c * LDK_ + r] = (_Float16)kvc_in[(bh * DH_ + r) * DH_ + c];
        kvsT[c * LDK_ + r] = (_Float16)kvs_in[(bh * DH_ + r) * DH_ + c];
    }
    if (tid < 64) {
        ksc[tid] = ksc_in[bh * DH_ + tid];
        kss[tid] = kss_in[bh * DH_ + tid];
    }
    __syncthreads();

    const int s0 = sbase + wave * 16;
    const int s  = s0 + lm;
    const float cw = cosw[b * S_ + s];
    const float sw = sinw[b * S_ + s];
    const _Float16* qrow = &Qp[(b * S_ + s) * D_ + hd * DH_];

    // Denominator (lanes 0..15 authoritative; 16..31 duplicate)
    float dn = 0.f;
    for (int d = 0; d < DH_; ++d) {
        const float qv = (float)qrow[d];
        dn += qv * (cw * ksc[d] + sw * kss[d]);
    }
    const float dinv = 1.0f / (dn + EPS_);

    v8f acc[4];
#pragma unroll
    for (int j = 0; j < 4; ++j) acc[j] = (v8f){0.f, 0.f, 0.f, 0.f, 0.f, 0.f, 0.f, 0.f};

#pragma unroll
    for (int kb = 0; kb < 2; ++kb) {
        v16h a_c, a_s;
#pragma unroll
        for (int h = 0; h < 16; ++h) {
            const int kk = kb * 32 + ((h < 8) ? (hi * 8 + h) : (16 + hi * 8 + (h - 8)));
            const float qv = (float)qrow[kk];
            a_c[h] = (_Float16)(qv * cw);
            a_s[h] = (_Float16)(qv * sw);
        }
#pragma unroll
        for (int j = 0; j < 4; ++j) {
            const v16h bc = frag_b(&kvcT[(j * 16 + lm) * LDK_ + kb * 32], hi);
            const v16h bs = frag_b(&kvsT[(j * 16 + lm) * LDK_ + kb * 32], hi);
            acc[j] = WMMA_F16(a_c, bc, acc[j]);
            acc[j] = WMMA_F16(a_s, bs, acc[j]);
        }
    }

    // Broadcast the 8 denominators once (ds_bpermute), reuse across 4 tiles
    float dmv[8];
#pragma unroll
    for (int r = 0; r < 8; ++r) dmv[r] = __shfl(dinv, hi * 8 + r, 32);

#pragma unroll
    for (int j = 0; j < 4; ++j) {
#pragma unroll
        for (int r = 0; r < 8; ++r) {
            const int rowM = hi * 8 + r;
            const float v  = acc[j][r] * dmv[r];
            attn[(b * S_ + s0 + rowM) * D_ + hd * DH_ + j * 16 + lm] = (_Float16)v;
        }
    }
}

// ---------------------------------------------------------------------------
// Host-side orchestration
// ---------------------------------------------------------------------------
extern "C" void kernel_launch(void* const* d_in, const int* in_sizes, int n_in,
                              void* d_out, int out_size, void* d_ws, size_t ws_size,
                              hipStream_t stream) {
    (void)in_sizes; (void)n_in; (void)out_size; (void)ws_size;

    const float*         X     = (const float*)d_in[0];
    const unsigned char* maskp = (const unsigned char*)d_in[1];
    const float* Wq = (const float*)d_in[2];
    const float* bq = (const float*)d_in[3];
    const float* Wk = (const float*)d_in[4];
    const float* bk = (const float*)d_in[5];
    const float* Wv = (const float*)d_in[6];
    const float* bv = (const float*)d_in[7];
    const float* Wo = (const float*)d_in[8];
    const float* bo = (const float*)d_in[9];
    float* out = (float*)d_out;

    // Workspace layout
    _Float16* Q    = (_Float16*)d_ws;
    _Float16* Kx   = Q  + (size_t)BS_ * D_;
    _Float16* V    = Kx + (size_t)BS_ * D_;
    _Float16* ATTN = V  + (size_t)BS_ * D_;
    float* KVC  = (float*)(ATTN + (size_t)BS_ * D_);
    float* KVS  = KVC + (size_t)B_ * H_ * DH_ * DH_;
    float* KSC  = KVS + (size_t)B_ * H_ * DH_ * DH_;
    float* KSS  = KSC + (size_t)B_ * H_ * DH_;
    float* COSW = KSS + (size_t)B_ * H_ * DH_;
    float* SINW = COSW + (size_t)B_ * S_;

    const int gemm_blocks = (BS_ / 128) * (D_ / 64);   // 2048

    cosformer_theta_kernel<<<B_, 256, 0, stream>>>(maskp, COSW, SINW);

    // Q = relu(X @ Wq + bq) ; K = relu(X @ Wk + bk) * mask ; V = X @ Wv + bv
    cosformer_gemm_kernel<false, true><<<gemm_blocks, 256, 0, stream>>>(X, Wq, bq, Q,  nullptr, 1);
    cosformer_gemm_kernel<false, true><<<gemm_blocks, 256, 0, stream>>>(X, Wk, bk, Kx, maskp,  1);
    cosformer_gemm_kernel<false, true><<<gemm_blocks, 256, 0, stream>>>(X, Wv, bv, V,  nullptr, 0);

    cosformer_kv_kernel<<<B_ * H_, 512, 0, stream>>>(Kx, V, COSW, SINW, KVC, KVS);
    cosformer_ksum_kernel<<<B_ * H_, 64, 0, stream>>>(Kx, COSW, SINW, KSC, KSS);
    cosformer_attn_kernel<<<B_ * H_ * (S_ / 128), 256, 0, stream>>>(
        Q, KVC, KVS, KSC, KSS, COSW, SINW, ATTN);

    // out = attn @ Wo + bo
    cosformer_gemm_kernel<true, false><<<gemm_blocks, 256, 0, stream>>>(ATTN, Wo, bo, out, nullptr, 0);
}
